// TrafficPredictor_35923106463788
// MI455X (gfx1250) — compile-verified, matrix operand order checked by reading, not run on hardware
//
#include <hip/hip_runtime.h>
#include <hip/hip_bf16.h>

// ---------------------------------------------------------------------------
// Problem constants (match reference)
// ---------------------------------------------------------------------------
#define N_NODES   500
#define N_EDGES   2000
#define BATCH     128
#define IN_F      64
#define OUT_F     4
#define N_HID     1024
#define N_HID_GNN 256
#define TOT_NODES (BATCH * N_NODES)   // 64000
#define TOT_EDGES (BATCH * N_EDGES)   // 256000
#define H_ELEMS   (BATCH * N_EDGES * OUT_F) // 1,024,000

typedef __attribute__((ext_vector_type(16))) _Float16 v16h;
typedef __attribute__((ext_vector_type(8)))  float    v8f;

struct MlpW {
  const _Float16* w[5];
  const float*    b[5];
};

// ---------------------------------------------------------------------------
// WMMA helpers (CDNA5 16x16x32 f16 -> f32)
// ---------------------------------------------------------------------------
__device__ __forceinline__ v8f wmma_16x16x32(v16h a, v16h b, v8f c) {
  return __builtin_amdgcn_wmma_f32_16x16x32_f16(
      /*neg_a=*/false, a, /*neg_b=*/false, b,
      /*c_mod=*/(short)0, c, /*reuse_a=*/false, /*reuse_b=*/false);
}

// A-matrix 16x32 f16 fragment (row-major source, ld in elements).
// lane<16: M=lane,     halves 0..7 = K 0..7,  8..15 = K 16..23
// lane>=16: M=lane-16, halves 0..7 = K 8..15, 8..15 = K 24..31
__device__ __forceinline__ v16h load_a_frag(const _Float16* base, int ld,
                                            int row0, int kt, int lane) {
  int m  = row0 + (lane & 15);
  int kb = kt * 32 + ((lane >> 4) << 3);
  const _Float16* p = base + (size_t)m * ld + kb;
  v16h r;
  float4* r4 = reinterpret_cast<float4*>(&r);
  r4[0] = *reinterpret_cast<const float4*>(p);
  r4[1] = *reinterpret_cast<const float4*>(p + 16);
  return r;
}

// B fragments are pre-packed: per (kt,nt) tile, 32 lanes x 16 halves contiguous.
__device__ __forceinline__ v16h load_b_frag(const _Float16* Wp, int ntiles,
                                            int kt, int nt, int lane) {
  const _Float16* p = Wp + (((size_t)kt * ntiles + nt) * 32 + lane) * 16;
  return *reinterpret_cast<const v16h*>(p);
}

// One hidden MLP layer inside LDS: out[64xN] = relu(in[64xK] @ Wp + b), f16 out.
// 4 waves, wave w owns rows 16w..16w+15.
__device__ __forceinline__ void mlp_layer_hidden(const _Float16* in, int K,
                                                 _Float16* out, int N,
                                                 const _Float16* Wp,
                                                 const float* bias) {
  const int lane = threadIdx.x & 31;
  const int r0   = (threadIdx.x >> 5) * 16;
  const int ktiles = K >> 5;
  const int ntiles = N >> 4;
  const int n  = lane & 15;
  const int mo = (lane >> 4) << 3;
  for (int nt = 0; nt < ntiles; ++nt) {
    v8f acc = {};
    for (int kt = 0; kt < ktiles; ++kt) {
      v16h a = load_a_frag(in, K, r0, kt, lane);
      v16h b = load_b_frag(Wp, ntiles, kt, nt, lane);
      acc = wmma_16x16x32(a, b, acc);
    }
    const int col = nt * 16 + n;
    const float bv = bias[col];
#pragma unroll
    for (int j = 0; j < 8; ++j) {
      float v = acc[j] + bv;
      v = v > 0.0f ? v : 0.0f;
      out[(size_t)(r0 + mo + j) * N + col] = (_Float16)v;
    }
  }
}

// Final layer: N padded to 16, real N = 4, no ReLU. Writes f16 (out16) or f32
// (out32) at row stride ldo; gRow0 added for global destinations.
__device__ __forceinline__ void mlp_layer_final(const _Float16* in, int K,
                                                const _Float16* Wp,
                                                const float* bias, int gRow0,
                                                _Float16* out16, float* out32,
                                                int ldo) {
  const int lane = threadIdx.x & 31;
  const int r0   = (threadIdx.x >> 5) * 16;
  const int ktiles = K >> 5;
  v8f acc = {};
  for (int kt = 0; kt < ktiles; ++kt) {
    v16h a = load_a_frag(in, K, r0, kt, lane);
    v16h b = load_b_frag(Wp, 1, kt, 0, lane);
    acc = wmma_16x16x32(a, b, acc);
  }
  const int n  = lane & 15;
  const int mo = (lane >> 4) << 3;
  if (n < OUT_F) {
    const float bv = bias[n];
#pragma unroll
    for (int j = 0; j < 8; ++j) {
      float v = acc[j] + bv;
      size_t idx = (size_t)(gRow0 + r0 + mo + j) * ldo + n;
      if (out32) out32[idx] = v;
      else       out16[idx] = (_Float16)v;
    }
  }
}

// ---------------------------------------------------------------------------
// Weight packer: f32 [Ksrc x Nsrc] row-major -> f16 fragment-packed
// [Kp/32 x Np/16 x 32 lanes x 16 halves], zero-padded.
// ---------------------------------------------------------------------------
__global__ void pack_w_kernel(const float* __restrict__ W, _Float16* __restrict__ Wp,
                              int Ksrc, int Nsrc, int Kp, int Np) {
  const int idx = blockIdx.x * blockDim.x + threadIdx.x;
  const int total = (Kp >> 5) * (Np >> 4) * 32;
  if (idx >= total) return;
  const int lane = idx & 31;
  const int t = idx >> 5;
  const int ntiles = Np >> 4;
  const int nt = t % ntiles;
  const int kt = t / ntiles;
  const int ncol = nt * 16 + (lane & 15);
  v16h tmp;
#pragma unroll
  for (int h = 0; h < 16; ++h) {
    int k = kt * 32 + ((lane >> 4) << 3) + (h & 7) + ((h >> 3) << 4);
    float v = (k < Ksrc && ncol < Nsrc) ? W[(size_t)k * Nsrc + ncol] : 0.0f;
    tmp[h] = (_Float16)v;
  }
  *reinterpret_cast<v16h*>(Wp + (size_t)idx * 16) = tmp;
}

__global__ void f32_to_f16_kernel(const float* __restrict__ in,
                                  _Float16* __restrict__ out, int n) {
  int i = blockIdx.x * blockDim.x + threadIdx.x;
  if (i < n) out[i] = (_Float16)in[i];
}

__global__ void zero_f32_kernel(float* __restrict__ p, int n) {
  int i = blockIdx.x * blockDim.x + threadIdx.x;
  if (i < n) p[i] = 0.0f;
}

// scatter-sum edge messages onto destination nodes (thread per e*4+j)
__global__ void scatter_agg_kernel(const float* __restrict__ h_e,
                                   const int* __restrict__ dst,
                                   float* __restrict__ agg, int n4) {
  int i = blockIdx.x * blockDim.x + threadIdx.x;
  if (i < n4) {
    int e = i >> 2, j = i & 3;
    atomicAdd(&agg[(size_t)dst[e] * 4 + j], h_e[i]);
  }
}

// ---------------------------------------------------------------------------
// Extractor GEMM: out = act(A[128xK] @ Wp + b). 8 waves/block; block b handles
// column tile b, wave w handles row tile w. A is f16 in global (L2-resident).
// ---------------------------------------------------------------------------
__global__ __launch_bounds__(256) void gemm_ext_kernel(
    const _Float16* __restrict__ A, const _Float16* __restrict__ Wp,
    const float* __restrict__ bias, int K, int N,
    _Float16* __restrict__ out16, float* __restrict__ out32a,
    float* __restrict__ out32b, int relu) {
  const int lane = threadIdx.x & 31;
  const int mt   = threadIdx.x >> 5;   // 0..7 row tile
  const int nt   = blockIdx.x;         // column tile
  const int ktiles = K >> 5;
  const int ntiles = N >> 4;

  const int m   = mt * 16 + (lane & 15);
  const int kb0 = (lane >> 4) << 3;
  const _Float16* arow = A + (size_t)m * K + kb0;

  v8f acc = {};
  for (int kt = 0; kt < ktiles; ++kt) {
    v16h a;
    float4* a4 = reinterpret_cast<float4*>(&a);
    const _Float16* p = arow + kt * 32;
    a4[0] = *reinterpret_cast<const float4*>(p);
    a4[1] = *reinterpret_cast<const float4*>(p + 16);
    v16h b = load_b_frag(Wp, ntiles, kt, nt, lane);
    acc = wmma_16x16x32(a, b, acc);
  }

  const int n   = lane & 15;
  const int mo  = (lane >> 4) << 3;
  const int col = nt * 16 + n;
  const float bv = bias[col];
#pragma unroll
  for (int j = 0; j < 8; ++j) {
    int row = mt * 16 + mo + j;
    float v = acc[j] + bv;
    if (relu) v = v > 0.0f ? v : 0.0f;
    size_t idx = (size_t)row * N + col;
    if (out16)  out16[idx]  = (_Float16)v;
    if (out32a) out32a[idx] = v;
    if (out32b) out32b[idx] = v;
  }
}

// ---------------------------------------------------------------------------
// Fused GNN node kernel: per block of 64 nodes, run reduce-MLP then node-MLP
// entirely in LDS. Input = [agg(4) | feat(64) | pad] -> 96.
// ---------------------------------------------------------------------------
__global__ __launch_bounds__(128) void gnn_node_kernel(
    const float* __restrict__ agg, const _Float16* __restrict__ feat,
    _Float16* __restrict__ nodeH, MlpW red, MlpW node) {
  __shared__ _Float16 sA[64 * 256];
  __shared__ _Float16 sB[64 * 256];
  const int row0 = blockIdx.x * 64;

  for (int i = threadIdx.x; i < 64 * 96; i += 128) {
    int r = i / 96, c = i % 96;
    int gr = row0 + r;
    _Float16 v = (_Float16)0.0f;
    if (c < 4)       v = (_Float16)agg[(size_t)gr * 4 + c];
    else if (c < 68) v = feat[(gr / N_NODES) * IN_F + (c - 4)];
    sA[i] = v;
  }
  __syncthreads();

  mlp_layer_hidden(sA, 96,  sB, 256, red.w[0], red.b[0]); __syncthreads();
  mlp_layer_hidden(sB, 256, sA, 256, red.w[1], red.b[1]); __syncthreads();
  mlp_layer_hidden(sA, 256, sB, 256, red.w[2], red.b[2]); __syncthreads();
  mlp_layer_hidden(sB, 256, sA, 256, red.w[3], red.b[3]); __syncthreads();
  // red final -> cols 0..3 of new 96-wide input in sB (no relu)
  mlp_layer_final(sA, 256, red.w[4], red.b[4], /*gRow0=*/0, sB, nullptr, 96);
  // refill feature / pad columns of sB
  for (int i = threadIdx.x; i < 64 * 96; i += 128) {
    int r = i / 96, c = i % 96;
    if (c >= 4) {
      int gr = row0 + r;
      _Float16 v = (_Float16)0.0f;
      if (c < 68) v = feat[(gr / N_NODES) * IN_F + (c - 4)];
      sB[i] = v;
    }
  }
  __syncthreads();

  mlp_layer_hidden(sB, 96,  sA, 256, node.w[0], node.b[0]); __syncthreads();
  mlp_layer_hidden(sA, 256, sB, 256, node.w[1], node.b[1]); __syncthreads();
  mlp_layer_hidden(sB, 256, sA, 256, node.w[2], node.b[2]); __syncthreads();
  mlp_layer_hidden(sA, 256, sB, 256, node.w[3], node.b[3]); __syncthreads();
  mlp_layer_final(sB, 256, node.w[4], node.b[4], row0, nodeH, nullptr, OUT_F);
}

// ---------------------------------------------------------------------------
// Fused GNN edge kernel: per block of 64 edges, update h_e in place.
// Input = [h_e(4) | node_h[src](4) | feat(64) | pad] -> 96.
// ---------------------------------------------------------------------------
__global__ __launch_bounds__(128) void gnn_edge_kernel(
    float* __restrict__ h_e, const _Float16* __restrict__ nodeH,
    const int* __restrict__ src, const _Float16* __restrict__ feat,
    MlpW edge) {
  __shared__ _Float16 sA[64 * 256];
  __shared__ _Float16 sB[64 * 256];
  const int row0 = blockIdx.x * 64;

  for (int i = threadIdx.x; i < 64 * 96; i += 128) {
    int r = i / 96, c = i % 96;
    int gr = row0 + r;
    _Float16 v = (_Float16)0.0f;
    if (c < 4)       v = (_Float16)h_e[(size_t)gr * 4 + c];
    else if (c < 8)  v = nodeH[(size_t)src[gr] * 4 + (c - 4)];
    else if (c < 72) v = feat[(gr / N_EDGES) * IN_F + (c - 8)];
    sA[i] = v;
  }
  __syncthreads();

  mlp_layer_hidden(sA, 96,  sB, 256, edge.w[0], edge.b[0]); __syncthreads();
  mlp_layer_hidden(sB, 256, sA, 256, edge.w[1], edge.b[1]); __syncthreads();
  mlp_layer_hidden(sA, 256, sB, 256, edge.w[2], edge.b[2]); __syncthreads();
  mlp_layer_hidden(sB, 256, sA, 256, edge.w[3], edge.b[3]); __syncthreads();
  mlp_layer_final(sA, 256, edge.w[4], edge.b[4], row0, nullptr, h_e, OUT_F);
}

// ---------------------------------------------------------------------------
// Host orchestration
// ---------------------------------------------------------------------------
extern "C" void kernel_launch(void* const* d_in, const int* in_sizes, int n_in,
                              void* d_out, int out_size, void* d_ws, size_t ws_size,
                              hipStream_t stream) {
  const float* features = (const float*)d_in[0];
  const int*   src      = (const int*)d_in[1];
  const int*   dst      = (const int*)d_in[2];
  const float *extW[9], *extB[9];
  for (int i = 0; i < 9; ++i) { extW[i] = (const float*)d_in[3 + i];  extB[i] = (const float*)d_in[12 + i]; }
  const float *edgW[5], *edgB[5], *nodW[5], *nodB[5], *redW[5], *redB[5];
  for (int i = 0; i < 5; ++i) {
    edgW[i] = (const float*)d_in[21 + i]; edgB[i] = (const float*)d_in[26 + i];
    nodW[i] = (const float*)d_in[31 + i]; nodB[i] = (const float*)d_in[36 + i];
    redW[i] = (const float*)d_in[41 + i]; redB[i] = (const float*)d_in[46 + i];
  }

  // --- workspace carve-out (all chunks 32B-aligned by construction) ---
  char* wsp = (char*)d_ws;
  auto carve = [&](size_t bytes) { void* p = (void*)wsp; wsp += bytes; return p; };

  _Float16* featF16 = (_Float16*)carve((size_t)BATCH * IN_F * 2);

  static const int extD[10] = {64,1024,1024,1024,1024,1024,1024,1024,1024,8000};
  _Float16* extWp[9];
  for (int i = 0; i < 9; ++i)
    extWp[i] = (_Float16*)carve((size_t)extD[i] * extD[i + 1] * 2);

  static const int gKp[5] = {96, 256, 256, 256, 256};   // padded K
  static const int gNp[5] = {256, 256, 256, 256, 16};   // padded N
  static const int eKs[5] = {72, 256, 256, 256, 256};   // real K (edge)
  static const int nKs[5] = {68, 256, 256, 256, 256};   // real K (node/red)
  static const int gNs[5] = {256, 256, 256, 256, 4};    // real N
  _Float16 *edgWp[5], *nodWp[5], *redWp[5];
  for (int i = 0; i < 5; ++i) edgWp[i] = (_Float16*)carve((size_t)gKp[i] * gNp[i] * 2);
  for (int i = 0; i < 5; ++i) nodWp[i] = (_Float16*)carve((size_t)gKp[i] * gNp[i] * 2);
  for (int i = 0; i < 5; ++i) redWp[i] = (_Float16*)carve((size_t)gKp[i] * gNp[i] * 2);

  _Float16* act0  = (_Float16*)carve((size_t)BATCH * N_HID * 2);
  _Float16* act1  = (_Float16*)carve((size_t)BATCH * N_HID * 2);
  float*    agg   = (float*)carve((size_t)TOT_NODES * 4 * 4);
  _Float16* nodeH = (_Float16*)carve((size_t)TOT_NODES * 4 * 2);

  float* h_out = (float*)d_out;            // [128, 8000]
  float* h_e   = (float*)d_out + H_ELEMS;  // [256000, 4], working + final

  // --- 1. convert features, pack all weights to f16 fragment layout ---
  f32_to_f16_kernel<<<(BATCH * IN_F + 255) / 256, 256, 0, stream>>>(
      features, featF16, BATCH * IN_F);
  for (int i = 0; i < 9; ++i) {
    int tot = (extD[i] >> 5) * (extD[i + 1] >> 4) * 32;
    pack_w_kernel<<<(tot + 255) / 256, 256, 0, stream>>>(
        extW[i], extWp[i], extD[i], extD[i + 1], extD[i], extD[i + 1]);
  }
  for (int i = 0; i < 5; ++i) {
    int tot = (gKp[i] >> 5) * (gNp[i] >> 4) * 32;
    pack_w_kernel<<<(tot + 255) / 256, 256, 0, stream>>>(
        edgW[i], edgWp[i], eKs[i], gNs[i], gKp[i], gNp[i]);
    pack_w_kernel<<<(tot + 255) / 256, 256, 0, stream>>>(
        nodW[i], nodWp[i], nKs[i], gNs[i], gKp[i], gNp[i]);
    pack_w_kernel<<<(tot + 255) / 256, 256, 0, stream>>>(
        redW[i], redWp[i], nKs[i], gNs[i], gKp[i], gNp[i]);
  }

  // --- 2. feature extractor (9 GEMM layers) ---
  const _Float16* Ain = featF16;
  _Float16* Aout = act0;
  for (int i = 0; i < 8; ++i) {
    int K = extD[i], N = extD[i + 1];
    gemm_ext_kernel<<<N / 16, 256, 0, stream>>>(
        Ain, extWp[i], extB[i], K, N, Aout, nullptr, nullptr, 1);
    Ain = Aout;
    Aout = (Aout == act0) ? act1 : act0;
  }
  // final layer: fp32 out to h AND the initial h_e (identity reshape)
  gemm_ext_kernel<<<8000 / 16, 256, 0, stream>>>(
      Ain, extWp[8], extB[8], 1024, 8000, nullptr, h_out, h_e, 0);

  // --- 3. GNN layers (weight-shared, 4 iterations) ---
  MlpW redM, nodM, edgM;
  for (int i = 0; i < 5; ++i) {
    redM.w[i] = redWp[i]; redM.b[i] = redB[i];
    nodM.w[i] = nodWp[i]; nodM.b[i] = nodB[i];
    edgM.w[i] = edgWp[i]; edgM.b[i] = edgB[i];
  }
  for (int l = 0; l < 4; ++l) {
    zero_f32_kernel<<<(TOT_NODES * 4 + 255) / 256, 256, 0, stream>>>(
        agg, TOT_NODES * 4);
    scatter_agg_kernel<<<(TOT_EDGES * 4 + 255) / 256, 256, 0, stream>>>(
        h_e, dst, agg, TOT_EDGES * 4);
    gnn_node_kernel<<<TOT_NODES / 64, 128, 0, stream>>>(
        agg, featF16, nodeH, redM, nodM);
    gnn_edge_kernel<<<TOT_EDGES / 64, 128, 0, stream>>>(
        h_e, nodeH, src, featF16, edgM);
  }
}
